// CaptionModel_61795989455377
// MI455X (gfx1250) — matrix operand order repaired and checked by reading, not compiled, not run on hardware
//
#include <hip/hip_runtime.h>
#include <stdint.h>

// Problem constants (from reference)
#define B_SZ 128
#define D_SZ 512
#define E_SZ 2048
#define V_SZ 20000
#define T_SZ 26
#define START_TOK 1
#define STOP_TOK 2

typedef __attribute__((ext_vector_type(16))) __bf16 v16bf;
typedef __attribute__((ext_vector_type(8)))  float  v8f;

// ---------------------------------------------------------------------------
// bf16 conversion (round-to-nearest-even)
// ---------------------------------------------------------------------------
__device__ __forceinline__ uint16_t f32_to_bf16(float f) {
  uint32_t u = __float_as_uint(f);
  u += 0x7FFFu + ((u >> 16) & 1u);
  return (uint16_t)(u >> 16);
}

// ---------------------------------------------------------------------------
// WMMA fragment packing (CDNA5 ISA 7.12.2, 16-bit operands, wave32)
// A 16x32 tile: lane = (m%16) + 16*half; vgpr v = 4*(kk>=16) + (kr&7)/2
// Per (tile, lane): 8 dwords = 32 bytes, contiguous -> b128-pair loads.
// ---------------------------------------------------------------------------
__device__ __forceinline__ void a_pack_store(uint16_t* base, int m, int k,
                                             int kc_off, int kc_stride, uint16_t v) {
  int kc = (k >> 5) + kc_off;
  int kk = k & 31;
  int grp = kk >> 4, kr = kk & 15, half = kr >> 3, rr = kr & 7;
  int vv = (grp << 2) + (rr >> 1), lo = rr & 1;
  int lane = (m & 15) + (half << 4), mb = m >> 4;
  size_t off = ((((size_t)mb * kc_stride + kc) * 32 + lane) * 8 + vv) * 2 + lo;
  base[off] = v;
}

// B 32x16 tile: lane = (n%16) + 16*((k%16) >= 8); vgpr v = 4*(k%32 >= 16) + (k%8)/2
__device__ __forceinline__ void b_pack_store(uint16_t* base, int k, int n,
                                             int KC, uint16_t v) {
  int nb = n >> 4, kc = k >> 5, kk = k & 31;
  int half = (kk >> 3) & 1, kr = ((kk >> 4) << 3) + (kk & 7), vv = kr >> 1, lo = kr & 1;
  int lane = (n & 15) + (half << 4);
  size_t off = ((((size_t)nb * KC + kc) * 32 + lane) * 8 + vv) * 2 + lo;
  base[off] = v;
}

// ---------------------------------------------------------------------------
// Packing kernels (run once per launch; weights stay bf16-packed in L2)
// ---------------------------------------------------------------------------
__global__ void pack_b_kernel(const float* __restrict__ src, uint16_t* __restrict__ dst,
                              int K, int N) {
  size_t id = (size_t)blockIdx.x * 256 + threadIdx.x;
  if (id >= (size_t)K * N) return;
  int k = (int)(id / N), n = (int)(id % N);
  b_pack_store(dst, k, n, K >> 5, f32_to_bf16(src[id]));
}

// Fused gate weight: Wcat[1024,2048] = [w_ih ; w_hh]
__global__ void pack_wcat_kernel(const float* __restrict__ w_ih,
                                 const float* __restrict__ w_hh,
                                 uint16_t* __restrict__ dst) {
  size_t id = (size_t)blockIdx.x * 256 + threadIdx.x;
  if (id >= (size_t)1024 * 2048) return;
  int k = (int)(id >> 11), n = (int)(id & 2047);
  float v = (k < 512) ? w_ih[(size_t)k * 2048 + n] : w_hh[(size_t)(k - 512) * 2048 + n];
  b_pack_store(dst, k, n, 32, f32_to_bf16(v));
}

// Pack a [128,K] f32 activation into A-fragment layout
__global__ void pack_a_kernel(const float* __restrict__ src, uint16_t* __restrict__ dst,
                              int K, int kc_stride, int kc_off) {
  size_t id = (size_t)blockIdx.x * 256 + threadIdx.x;
  if (id >= (size_t)B_SZ * K) return;
  int m = (int)(id / K), k = (int)(id % K);
  a_pack_store(dst, m, k, kc_off, kc_stride, f32_to_bf16(src[id]));
}

// inp_0 = broadcast emb[START] into A rows 0..127 (kc 0..15 of gatesA); zero stopped
__global__ void init_inp_kernel(const float* __restrict__ emb, uint16_t* __restrict__ gatesA,
                                int* __restrict__ stopped) {
  int id = blockIdx.x * 256 + threadIdx.x;   // 128*512 threads
  int m = id >> 9, k = id & 511;
  a_pack_store(gatesA, m, k, 0, 32, f32_to_bf16(emb[(size_t)START_TOK * D_SZ + k]));
  if (id < B_SZ) stopped[id] = 0;
}

// ---------------------------------------------------------------------------
// WMMA GEMM: C[128,N] = A[128,K]*B[K,N] + bias.  Block = 8 waves = 8 M-tiles
// (EXEC always full for WMMA). Each wave: 4 N-tiles per A-fragment (reuse A).
// Software-pipelined with 2x K-unroll ping-pong buffers: buffer copies are
// implicit in the unroll (no v_mov rotation), loads overlap the matrix pipe
// via partial s_wait_loadcnt. KC is always even (16/32/64).
// ---------------------------------------------------------------------------
__global__ __launch_bounds__(256) void wmma_gemm_kernel(
    const uint16_t* __restrict__ Apack, const uint16_t* __restrict__ Bpack,
    const float* __restrict__ bias, float* __restrict__ out,
    int n_tiles, int KC, int a_kc_stride, int a_kc_off, int row_stride) {
  const int lane = threadIdx.x & 31;
  const int mb   = threadIdx.x >> 5;   // 8 waves -> M tiles 0..7 (M = 128)
  const int nb0  = blockIdx.x << 2;    // 4 N tiles per wave

  const char* Ab = (const char*)Apack;
  const char* Bb = (const char*)Bpack;

  const size_t a_base = (((size_t)(mb * a_kc_stride + a_kc_off)) * 32 + (size_t)lane) << 5;
  size_t b_base[4];
#pragma unroll
  for (int j = 0; j < 4; ++j) {
    int nb = nb0 + j;
    if (nb >= n_tiles) nb = n_tiles - 1;          // clamp: keep EXEC full, skip store later
    b_base[j] = (((size_t)nb * KC) * 32 + (size_t)lane) << 5;
  }

  const v8f zero = {0.f, 0.f, 0.f, 0.f, 0.f, 0.f, 0.f, 0.f};
  v8f acc[4];
#pragma unroll
  for (int j = 0; j < 4; ++j) acc[j] = zero;

  // ---- prologue: buffer 0 <- kc = 0 ----
  v16bf a0 = *(const v16bf*)(Ab + a_base);
  v16bf b0[4];
#pragma unroll
  for (int j = 0; j < 4; ++j) b0[j] = *(const v16bf*)(Bb + b_base[j]);

  v16bf a1;
  v16bf b1[4];

  // ---- steady state: two K-chunks per iteration, ping-pong buffers ----
#pragma unroll 1
  for (int kc = 0; kc < KC - 2; kc += 2) {
    const size_t k1 = (size_t)(kc + 1) << 10;     // fragment stride = 1 KB
    a1 = *(const v16bf*)(Ab + a_base + k1);
#pragma unroll
    for (int j = 0; j < 4; ++j) b1[j] = *(const v16bf*)(Bb + b_base[j] + k1);

#pragma unroll
    for (int j = 0; j < 4; ++j)
      acc[j] = __builtin_amdgcn_wmma_f32_16x16x32_bf16(
          false, a0, false, b0[j], (short)0, acc[j], false, false);

    const size_t k2 = (size_t)(kc + 2) << 10;
    a0 = *(const v16bf*)(Ab + a_base + k2);
#pragma unroll
    for (int j = 0; j < 4; ++j) b0[j] = *(const v16bf*)(Bb + b_base[j] + k2);

#pragma unroll
    for (int j = 0; j < 4; ++j)
      acc[j] = __builtin_amdgcn_wmma_f32_16x16x32_bf16(
          false, a1, false, b1[j], (short)0, acc[j], false, false);
  }

  // ---- tail: kc = KC-2 (in buf0) and kc = KC-1 ----
  {
    const size_t kl = (size_t)(KC - 1) << 10;
    a1 = *(const v16bf*)(Ab + a_base + kl);
#pragma unroll
    for (int j = 0; j < 4; ++j) b1[j] = *(const v16bf*)(Bb + b_base[j] + kl);

#pragma unroll
    for (int j = 0; j < 4; ++j)
      acc[j] = __builtin_amdgcn_wmma_f32_16x16x32_bf16(
          false, a0, false, b0[j], (short)0, acc[j], false, false);
#pragma unroll
    for (int j = 0; j < 4; ++j)
      acc[j] = __builtin_amdgcn_wmma_f32_16x16x32_bf16(
          false, a1, false, b1[j], (short)0, acc[j], false, false);
  }

  // C layout: vgpr r -> M = mb*16 + r + 8*(lane>=16); N = nb*16 + (lane&15)
  const int ncol  = lane & 15;
  const int mrow0 = (mb << 4) + ((lane >> 4) << 3);
#pragma unroll
  for (int j = 0; j < 4; ++j) {
    if (nb0 + j < n_tiles) {
      int n = ((nb0 + j) << 4) + ncol;
      float bv = bias[n];
#pragma unroll
      for (int r = 0; r < 8; ++r)
        out[(size_t)(mrow0 + r) * row_stride + n] = acc[j][r] + bv;
    }
  }
}

// ---------------------------------------------------------------------------
// LSTM pointwise: c' = sig(f)*c + sig(i)*tanh(g); h' = sig(o)*tanh(c')
// Writes h' directly as packed bf16 A-fragments (kc 16..31 of gatesA), which
// feeds BOTH next gates GEMM and this step's fc GEMM. PyTorch gate order ifgo.
// ---------------------------------------------------------------------------
__global__ void lstm_pointwise_kernel(const float* __restrict__ gates,
                                      float* __restrict__ c,
                                      uint16_t* __restrict__ gatesA) {
  int id = blockIdx.x * 256 + threadIdx.x;  // 128*512 threads
  int bi = id >> 9, d = id & 511;
  const float* g = gates + ((size_t)bi << 11);
  float ig = 1.f / (1.f + __expf(-g[d]));
  float fg = 1.f / (1.f + __expf(-g[d + 512]));
  float gg = tanhf(g[d + 1024]);
  float og = 1.f / (1.f + __expf(-g[d + 1536]));
  float c2 = fg * c[id] + ig * gg;
  float h2 = og * tanhf(c2);
  c[id] = c2;
  a_pack_store(gatesA, bi, d, /*kc_off=*/16, /*kc_stride=*/32, f32_to_bf16(h2));
}

// ---------------------------------------------------------------------------
// Per-row argmax over V (first-index tie-break), token freeze, emb gather+pack
// ---------------------------------------------------------------------------
__global__ __launch_bounds__(256) void argmax_gather_kernel(
    const float* __restrict__ out_base, float* __restrict__ tok_out, int t,
    const float* __restrict__ emb, uint16_t* __restrict__ gatesA,
    int* __restrict__ stopped) {
  __shared__ float sv[256];
  __shared__ int   si[256];
  __shared__ int   sidx;
  int b = blockIdx.x, tid = threadIdx.x;
  const float* row = out_base + (size_t)b * T_SZ * V_SZ + (size_t)t * V_SZ;

  float best = -3.402823466e38f;
  int bidx = 0x7fffffff;
  for (int i = tid; i < V_SZ; i += 256) {
    float v = row[i];
    if (v > best) { best = v; bidx = i; }   // strict > keeps lowest index per thread
  }
  sv[tid] = best; si[tid] = bidx;
  __syncthreads();
  for (int s = 128; s > 0; s >>= 1) {
    if (tid < s) {
      float v2 = sv[tid + s]; int i2 = si[tid + s];
      if (v2 > sv[tid] || (v2 == sv[tid] && i2 < si[tid])) { sv[tid] = v2; si[tid] = i2; }
    }
    __syncthreads();
  }
  if (tid == 0) {
    int idx = si[0];
    sidx = idx;
    int s = stopped[b];
    tok_out[(size_t)b * T_SZ + t] = (float)(s ? STOP_TOK : idx);  // freeze after STOP
    stopped[b] = s | (idx == STOP_TOK);
  }
  __syncthreads();
  int idx = sidx;
  // next inp = emb[idx] -> packed bf16 A-fragments (kc 0..15 of gatesA), row m=b
  for (int k = tid; k < D_SZ; k += 256)
    a_pack_store(gatesA, b, k, 0, 32, f32_to_bf16(emb[(size_t)idx * D_SZ + k]));
}

// ---------------------------------------------------------------------------
// Host orchestration (graph-capture safe: only kernel launches on `stream`)
// ---------------------------------------------------------------------------
extern "C" void kernel_launch(void* const* d_in, const int* in_sizes, int n_in,
                              void* d_out, int out_size, void* d_ws, size_t ws_size,
                              hipStream_t stream) {
  (void)in_sizes; (void)n_in; (void)out_size; (void)ws_size;

  const float* x    = (const float*)d_in[0];   // [128,2048]
  const float* emb  = (const float*)d_in[1];   // [20000,512]
  const float* w_ih = (const float*)d_in[2];   // [512,2048]
  const float* w_hh = (const float*)d_in[3];   // [512,2048]
  const float* bg   = (const float*)d_in[4];   // [2048]
  const float* fc_w = (const float*)d_in[5];   // [512,20000]
  const float* fc_b = (const float*)d_in[6];   // [20000]
  const float* ih_w = (const float*)d_in[7];   // [2048,512]
  const float* ih_b = (const float*)d_in[8];   // [512]
  const float* ic_w = (const float*)d_in[9];   // [2048,512]
  const float* ic_b = (const float*)d_in[10];  // [512]
  float* out = (float*)d_out;                  // logits [128,26,20000] ++ toks [128,26]

  // bump allocator over workspace (256 B aligned)
  char* ws = (char*)d_ws;
  size_t cur = 0;
  auto alloc = [&](size_t bytes) -> char* {
    size_t o = (cur + 255) & ~(size_t)255;
    cur = o + bytes;
    return ws + o;
  };

  uint16_t* fcW    = (uint16_t*)alloc((size_t)(V_SZ / 16) * (D_SZ / 32) * 1024);  // 20.48 MB
  uint16_t* Wcat   = (uint16_t*)alloc((size_t)(2048 / 16) * (1024 / 32) * 1024);  // 4 MB
  uint16_t* ihW    = (uint16_t*)alloc((size_t)(512 / 16) * (2048 / 32) * 1024);   // 2 MB
  uint16_t* icW    = (uint16_t*)alloc((size_t)(512 / 16) * (2048 / 32) * 1024);   // 2 MB
  uint16_t* xA     = (uint16_t*)alloc((size_t)8 * 64 * 1024);                     // 512 KB
  uint16_t* gatesA = (uint16_t*)alloc((size_t)8 * 32 * 1024);  // [inp | h] bf16  // 256 KB
  float*    gates  = (float*)alloc((size_t)B_SZ * 2048 * 4);                      // 1 MB
  float*    cbuf   = (float*)alloc((size_t)B_SZ * D_SZ * 4);
  float*    htmp   = (float*)alloc((size_t)B_SZ * D_SZ * 4);
  int*      stop_f = (int*)alloc(B_SZ * 4);

  // ---- one-time (per launch) bf16 conversion + fragment pre-swizzle ----
  pack_b_kernel<<<(512 * 20000) / 256, 256, 0, stream>>>(fc_w, fcW, 512, 20000);
  pack_wcat_kernel<<<(1024 * 2048) / 256, 256, 0, stream>>>(w_ih, w_hh, Wcat);
  pack_b_kernel<<<(2048 * 512) / 256, 256, 0, stream>>>(ih_w, ihW, 2048, 512);
  pack_b_kernel<<<(2048 * 512) / 256, 256, 0, stream>>>(ic_w, icW, 2048, 512);
  pack_a_kernel<<<(128 * 2048) / 256, 256, 0, stream>>>(x, xA, 2048, 64, 0);
  init_inp_kernel<<<(128 * 512) / 256, 256, 0, stream>>>(emb, gatesA, stop_f);

  // ---- init states: h0 = x@ih_w + ih_b ; c0 = x@ic_w + ic_b (K=2048, N=512) ----
  wmma_gemm_kernel<<<8, 256, 0, stream>>>(xA, ihW, ih_b, htmp, 32, 64, 64, 0, 512);
  wmma_gemm_kernel<<<8, 256, 0, stream>>>(xA, icW, ic_b, cbuf, 32, 64, 64, 0, 512);
  pack_a_kernel<<<(128 * 512) / 256, 256, 0, stream>>>(htmp, gatesA, 512, 32, 16);

  // ---- T sequential decode steps ----
  for (int t = 0; t < T_SZ; ++t) {
    // gates = [inp,h] @ [w_ih;w_hh] + b : K=1024 (KC=32), N=2048 (128 tiles)
    wmma_gemm_kernel<<<32, 256, 0, stream>>>(gatesA, Wcat, bg, gates,
                                             128, 32, 32, 0, 2048);
    // c,h update; h' packed into gatesA kc 16..31
    lstm_pointwise_kernel<<<(128 * 512) / 256, 256, 0, stream>>>(gates, cbuf, gatesA);
    // logits = h' @ fc_w + fc_b : K=512 (KC=16), N=20000 (1250 tiles, 313 groups)
    wmma_gemm_kernel<<<313, 256, 0, stream>>>(gatesA, fcW, fc_b,
                                              out + (size_t)t * V_SZ,
                                              1250, 16, 32, 16, T_SZ * V_SZ);
    // greedy token, freeze logic, gather emb[idx] -> gatesA kc 0..15
    argmax_gather_kernel<<<128, 256, 0, stream>>>(out,
                                                  out + (size_t)B_SZ * T_SZ * V_SZ,
                                                  t, emb, gatesA, stop_f);
  }
}
